// EssentialMemory_36790689857542
// MI455X (gfx1250) — compile-verified, matrix operand order checked by reading, not run on hardware
//
#include <hip/hip_runtime.h>
#include <math.h>

// ---------------------------------------------------------------------------
// EssentialMemory pipeline for MI455X (gfx1250, wave32, WMMA bf16).
//
// Sizes (fixed by reference): B=1024, D=512, Na=32768, Ns=65536, topk=64.
// Outputs (flat f32, concat): output[1024*512] | basin_ids[1024] |
//                             attn[1024*32768] | new_last_activated[65536]
//
// All GEMM operands are pre-converted to bf16 so the WMMA K-loop is pure
// global_load_b128 + v_wmma_f32_16x16x32_bf16 (no in-loop cvt VALU).
// ---------------------------------------------------------------------------

typedef __attribute__((ext_vector_type(16))) __bf16 v16bf;
typedef __attribute__((ext_vector_type(8)))  float  v8f;

#define D_MODEL  512
#define B_ROWS   1024
#define NACT     32768
#define NSLOT    65536
#define NOW_F    1.7e9f

__device__ __forceinline__ unsigned short f2bf(float f) {
  unsigned u = __float_as_uint(f);
  unsigned r = u + 0x7FFFu + ((u >> 16) & 1u);   // round-to-nearest-even
  return (unsigned short)(r >> 16);
}
__device__ __forceinline__ float bf2f(unsigned short h) {
  return __uint_as_float(((unsigned)h) << 16);
}
// monotonic float->uint key: larger float => larger key
__device__ __forceinline__ unsigned fkey(float f) {
  unsigned u = __float_as_uint(f);
  return (u & 0x80000000u) ? ~u : (u | 0x80000000u);
}

union FragBF {
  v16bf v;
  unsigned short u[16];
  uint4 q[2];
};

// ---------------------------------------------------------------------------
// Tiled NT GEMM: C[m,n] = scale * colscale[n] * sum_k A[m,k] * B[n,k]
// A, B: bf16 row-major, contraction on fast axis; K-step 32 via WMMA bf16.
// Block = 128 threads = 4 waves; 64x64 tile; 32x32 per wave (2x2 WMMA tiles).
// M, N multiples of 64; Kd multiple of 64.
// 16-bit A/B fragment layout (wave32): lane l holds row (l&15), two 16-byte
// K-chunks at K = base and K = base+16 with base = (l<16 ? 0 : 8).
// ---------------------------------------------------------------------------
template <bool HAS_CF, bool HAS_CBF, bool HAS_CS>
__global__ __launch_bounds__(128) void gemm_nt_wmma_kernel(
    const unsigned short* __restrict__ A, const unsigned short* __restrict__ Bw,
    float* __restrict__ Cf, unsigned short* __restrict__ Cbf,
    int M, int N, int Kd, float scale, const float* __restrict__ colscale)
{
  const int lane  = threadIdx.x & 31;
  const int wave  = threadIdx.x >> 5;
  const int bm    = blockIdx.y * 64;
  const int bn    = blockIdx.x * 64;
  const int wm    = bm + (wave >> 1) * 32;
  const int wn    = bn + (wave & 1) * 32;
  const int rsel  = lane & 15;
  const int khalf = (lane >> 4) * 8;       // 0 or 8 per A/B 16-bit layout

  const unsigned short* aptr[2];
  const unsigned short* bptr[2];
#pragma unroll
  for (int mi = 0; mi < 2; ++mi)
    aptr[mi] = A + (size_t)(wm + mi * 16 + rsel) * Kd;
#pragma unroll
  for (int ni = 0; ni < 2; ++ni)
    bptr[ni] = Bw + (size_t)(wn + ni * 16 + rsel) * Kd;

  v8f c[2][2];
  const v8f zero = {0.f, 0.f, 0.f, 0.f, 0.f, 0.f, 0.f, 0.f};
  c[0][0] = zero; c[0][1] = zero; c[1][0] = zero; c[1][1] = zero;

#pragma unroll 2
  for (int k = 0; k < Kd; k += 32) {
    FragBF a[2], b[2];
#pragma unroll
    for (int mi = 0; mi < 2; ++mi) {
      const unsigned short* p = aptr[mi] + k + khalf;
      __builtin_prefetch(p + 64, 0, 1);                    // two K-steps ahead
      a[mi].q[0] = *(const uint4*)(p);
      a[mi].q[1] = *(const uint4*)(p + 16);
    }
#pragma unroll
    for (int ni = 0; ni < 2; ++ni) {
      const unsigned short* p = bptr[ni] + k + khalf;
      b[ni].q[0] = *(const uint4*)(p);
      b[ni].q[1] = *(const uint4*)(p + 16);
    }
#pragma unroll
    for (int mi = 0; mi < 2; ++mi)
#pragma unroll
      for (int ni = 0; ni < 2; ++ni)
        c[mi][ni] = __builtin_amdgcn_wmma_f32_16x16x32_bf16(
            false, a[mi].v, false, b[ni].v, (short)0, c[mi][ni], false, false);
  }

  // C/D layout: VGPR v -> row = tile_m + v + (lane>=16 ? 8 : 0), col = tile_n + (lane&15)
#pragma unroll
  for (int ni = 0; ni < 2; ++ni) {
    const int col = wn + ni * 16 + rsel;
    float cs = scale;
    if (HAS_CS) cs *= colscale[col];
#pragma unroll
    for (int mi = 0; mi < 2; ++mi) {
      const int rbase = wm + mi * 16 + khalf;
#pragma unroll
      for (int v = 0; v < 8; ++v) {
        float val = c[mi][ni][v] * cs;
        size_t o = (size_t)(rbase + v) * N + col;
        if (HAS_CF)  Cf[o]  = val;
        if (HAS_CBF) Cbf[o] = f2bf(val);
      }
    }
  }
}

// ---------------------------------------------------------------------------
// Small helpers
// ---------------------------------------------------------------------------
__global__ void cvt_bf16_kernel(const float* __restrict__ src,
                                unsigned short* __restrict__ dst, int n) {
  int i = blockIdx.x * blockDim.x + threadIdx.x;
  if (i < n) dst[i] = f2bf(src[i]);
}

// dst[r, :] = bf16(src[idx[r], :])  with cols == D_MODEL (512)
__global__ void gather_cvt_kernel(const float* __restrict__ src,
                                  const int* __restrict__ idx,
                                  unsigned short* __restrict__ dst, int total) {
  int i = blockIdx.x * blockDim.x + threadIdx.x;
  if (i < total) {
    int r = i >> 9;                 // /512
    int c = i & 511;
    dst[i] = f2bf(src[(size_t)idx[r] * D_MODEL + c]);
  }
}

__global__ void gather_conf_kernel(const float* __restrict__ conf,
                                   const int* __restrict__ idx,
                                   float* __restrict__ out, int n) {
  int i = blockIdx.x * blockDim.x + threadIdx.x;
  if (i < n) out[i] = conf[idx[i]];
}

__global__ void copy_f32_kernel(const float* __restrict__ src,
                                float* __restrict__ dst, int n) {
  int i = blockIdx.x * blockDim.x + threadIdx.x;
  if (i < n) dst[i] = src[i];
}

__global__ void scatter_now_kernel(const int* __restrict__ basin_i,
                                   float* __restrict__ out_last, int n) {
  int i = blockIdx.x * blockDim.x + threadIdx.x;
  if (i < n) out_last[basin_i[i]] = NOW_F;   // duplicates collapse, same value
}

// ---------------------------------------------------------------------------
// Per-row top-k mask + softmax, in place on the attn region.
// One block per row; row (128 KB) staged in dynamic LDS (320 KB/WGP on CDNA5).
// Threshold = exact 64th-largest via 4-pass byte radix select on monotone key.
// Also emits argmax -> basin id.
// ---------------------------------------------------------------------------
__global__ __launch_bounds__(256) void topk_softmax_kernel(
    float* __restrict__ attn, const int* __restrict__ active_idx,
    const int* __restrict__ topk_ptr, float* __restrict__ basin_f,
    int* __restrict__ basin_i, int N)
{
  extern __shared__ float srow[];            // N floats
  __shared__ unsigned hist[256];
  __shared__ float redf[256];
  __shared__ int   redi[256];
  __shared__ unsigned s_prefix;
  __shared__ int s_need;

  const int row = blockIdx.x;
  const int t   = threadIdx.x;
  float* S = attn + (size_t)row * N;

  for (int j = t; j < N; j += 256) srow[j] = S[j];
  __syncthreads();

  // row max + first-occurrence argmax
  float m = -3.4e38f; int mi = 0x7FFFFFFF;
  for (int j = t; j < N; j += 256) {
    float v = srow[j];
    if (v > m) { m = v; mi = j; }
  }
  redf[t] = m; redi[t] = mi;
  __syncthreads();
  for (int s = 128; s > 0; s >>= 1) {
    if (t < s) {
      float vo = redf[t + s]; int io = redi[t + s];
      if (vo > redf[t] || (vo == redf[t] && io < redi[t])) { redf[t] = vo; redi[t] = io; }
    }
    __syncthreads();
  }
  const float rowMax = redf[0];
  const int   rowArg = redi[0];

  const int  topk    = topk_ptr[0];
  const bool useTopk = (topk > 0) && (topk < N);
  unsigned thrKey = 0u;
  if (useTopk) {
    unsigned prefix = 0u; int need = topk;
    for (int shift = 24; shift >= 0; shift -= 8) {
      hist[t] = 0u;
      __syncthreads();
      const unsigned maskHigh = (shift == 24) ? 0u : (0xFFFFFFFFu << (shift + 8));
      for (int j = t; j < N; j += 256) {
        unsigned k = fkey(srow[j]);
        if ((k & maskHigh) == prefix)
          atomicAdd(&hist[(k >> shift) & 0xFFu], 1u);
      }
      __syncthreads();
      if (t == 0) {
        int acc = 0, sel = 0;
        for (int b = 255; b >= 0; --b) {
          acc += (int)hist[b];
          if (acc >= need) { sel = b; break; }
        }
        s_need   = need - (acc - (int)hist[sel]);
        s_prefix = prefix | ((unsigned)sel << shift);
      }
      __syncthreads();
      prefix = s_prefix; need = s_need;
      __syncthreads();
    }
    thrKey = prefix;
  }

  // softmax denominator over kept entries
  float sum = 0.f;
  for (int j = t; j < N; j += 256) {
    float v = srow[j];
    if (!useTopk || fkey(v) >= thrKey) sum += __expf(v - rowMax);
  }
  redf[t] = sum;
  __syncthreads();
  for (int s = 128; s > 0; s >>= 1) {
    if (t < s) redf[t] += redf[t + s];
    __syncthreads();
  }
  const float inv = 1.0f / redf[0];

  for (int j = t; j < N; j += 256) {
    float v = srow[j];
    S[j] = (!useTopk || fkey(v) >= thrKey) ? __expf(v - rowMax) * inv : 0.f;
  }
  if (t == 0) {
    int g = active_idx[rowArg];
    basin_f[row] = (float)g;
    basin_i[row] = g;
  }
}

// ---------------------------------------------------------------------------
// ctx[b,:] = sum_j attn[b,j] * Vp[j,:]  — attn has ~64 nonzeros per row.
// Deterministic ordered compaction (chunk counts + serial exclusive scan),
// then dense accumulation over the compact list in ascending-j order.
// Emits bf16 ctx (A-operand of the final WMMA GEMM).
// ---------------------------------------------------------------------------
__global__ __launch_bounds__(256) void ctx_sparse_kernel(
    const float* __restrict__ attn, const unsigned short* __restrict__ Vp,
    unsigned short* __restrict__ ctx_bf, int N, int D)
{
  extern __shared__ float srow[];            // N floats
  __shared__ int   cnt[257];
  __shared__ int   s_n;
  __shared__ int   lidx[256];
  __shared__ float lp[256];

  const int row = blockIdx.x;
  const int t   = threadIdx.x;
  const float* S = attn + (size_t)row * N;

  for (int j = t; j < N; j += 256) srow[j] = S[j];
  __syncthreads();

  const int chunk = N / 256;                 // 128
  int c = 0;
  for (int j = t * chunk; j < (t + 1) * chunk; ++j)
    if (srow[j] != 0.f) ++c;
  cnt[t] = c;
  __syncthreads();
  if (t == 0) {
    int acc = 0;
    for (int i = 0; i < 256; ++i) { int v = cnt[i]; cnt[i] = acc; acc += v; }
    cnt[256] = acc;
    s_n = acc > 256 ? 256 : acc;
  }
  __syncthreads();
  int off = cnt[t];
  for (int j = t * chunk; j < (t + 1) * chunk; ++j) {
    float v = srow[j];
    if (v != 0.f) {
      if (off < 256) { lidx[off] = j; lp[off] = v; }
      ++off;
    }
  }
  __syncthreads();

  const int n = s_n;
  float acc0 = 0.f, acc1 = 0.f;               // D=512, 256 threads -> 2 cols
  for (int i = 0; i < n; ++i) {
    float p = lp[i];
    const unsigned short* vr = Vp + (size_t)lidx[i] * D;
    acc0 += p * bf2f(vr[t]);
    acc1 += p * bf2f(vr[t + 256]);
  }
  ctx_bf[(size_t)row * D + t]       = f2bf(acc0);
  ctx_bf[(size_t)row * D + t + 256] = f2bf(acc1);
}

// ---------------------------------------------------------------------------
extern "C" void kernel_launch(void* const* d_in, const int* in_sizes, int n_in,
                              void* d_out, int out_size, void* d_ws, size_t ws_size,
                              hipStream_t stream) {
  const float* query      = (const float*)d_in[0];
  const int*   active_idx = (const int*)  d_in[1];
  const float* keys       = (const float*)d_in[2];
  const float* values     = (const float*)d_in[3];
  const float* confidence = (const float*)d_in[4];
  const float* last_act   = (const float*)d_in[5];
  const float* Wq         = (const float*)d_in[6];
  const float* Wk         = (const float*)d_in[7];
  const float* Wv         = (const float*)d_in[8];
  const float* Wo         = (const float*)d_in[9];
  const int*   topk_ptr   = (const int*)  d_in[10];

  // workspace carve-out (256B aligned)
  char* wsp = (char*)d_ws;
  auto carve = [&](size_t bytes) -> char* {
    char* p = wsp;
    wsp += (bytes + 255) & ~(size_t)255;
    return p;
  };
  const size_t WB  = (size_t)D_MODEL * D_MODEL * sizeof(unsigned short);
  const size_t GB2 = (size_t)NACT * D_MODEL * sizeof(unsigned short);
  unsigned short* Wq_b  = (unsigned short*)carve(WB);
  unsigned short* Wk_b  = (unsigned short*)carve(WB);
  unsigned short* Wv_b  = (unsigned short*)carve(WB);
  unsigned short* Wo_b  = (unsigned short*)carve(WB);
  unsigned short* Qb    = (unsigned short*)carve((size_t)B_ROWS * D_MODEL * sizeof(unsigned short));
  unsigned short* Kg    = (unsigned short*)carve(GB2);   // gathered keys, bf16
  unsigned short* Vg    = (unsigned short*)carve(GB2);   // gathered values, bf16
  float*          conf_a = (float*)carve((size_t)NACT * sizeof(float));
  unsigned short* Qp    = (unsigned short*)carve((size_t)B_ROWS * D_MODEL * sizeof(unsigned short));
  unsigned short* Kp    = (unsigned short*)carve(GB2);
  unsigned short* Vp    = (unsigned short*)carve(GB2);
  unsigned short* ctx_b = (unsigned short*)carve((size_t)B_ROWS * D_MODEL * sizeof(unsigned short));
  int*            basin_i = (int*)carve((size_t)B_ROWS * sizeof(int));

  // output regions (flat f32, concat in return order)
  float* out       = (float*)d_out;
  float* out_y     = out;                                       // 1024*512
  float* out_basin = out + (size_t)B_ROWS * D_MODEL;            // 1024
  float* out_attn  = out_basin + B_ROWS;                        // 1024*32768
  float* out_last  = out_attn + (size_t)B_ROWS * NACT;          // 65536

  // 1) one-time bf16 conversions + gathers (removes all in-GEMM cvt VALU)
  const int WN = D_MODEL * D_MODEL;
  cvt_bf16_kernel<<<(WN + 255) / 256, 256, 0, stream>>>(Wq, Wq_b, WN);
  cvt_bf16_kernel<<<(WN + 255) / 256, 256, 0, stream>>>(Wk, Wk_b, WN);
  cvt_bf16_kernel<<<(WN + 255) / 256, 256, 0, stream>>>(Wv, Wv_b, WN);
  cvt_bf16_kernel<<<(WN + 255) / 256, 256, 0, stream>>>(Wo, Wo_b, WN);
  const int QN = B_ROWS * D_MODEL;
  cvt_bf16_kernel<<<(QN + 255) / 256, 256, 0, stream>>>(query, Qb, QN);
  const int GN = NACT * D_MODEL;
  gather_cvt_kernel<<<(GN + 255) / 256, 256, 0, stream>>>(keys,   active_idx, Kg, GN);
  gather_cvt_kernel<<<(GN + 255) / 256, 256, 0, stream>>>(values, active_idx, Vg, GN);
  gather_conf_kernel<<<(NACT + 255) / 256, 256, 0, stream>>>(confidence, active_idx, conf_a, NACT);

  const dim3 blk(128);
  // 2) Q = query @ Wq^T  (bf16 out, A-side of scores GEMM)
  gemm_nt_wmma_kernel<false, true, false>
      <<<dim3(D_MODEL / 64, B_ROWS / 64), blk, 0, stream>>>(
      Qb, Wq_b, nullptr, Qp, B_ROWS, D_MODEL, D_MODEL, 1.0f, nullptr);
  // 3) K = keys[active_idx] @ Wk^T  (bf16 out, B-side of scores GEMM)
  gemm_nt_wmma_kernel<false, true, false>
      <<<dim3(D_MODEL / 64, NACT / 64), blk, 0, stream>>>(
      Kg, Wk_b, nullptr, Kp, NACT, D_MODEL, D_MODEL, 1.0f, nullptr);
  // 4) V = values[active_idx] @ Wv^T  (bf16 out, feeds sparse ctx)
  gemm_nt_wmma_kernel<false, true, false>
      <<<dim3(D_MODEL / 64, NACT / 64), blk, 0, stream>>>(
      Vg, Wv_b, nullptr, Vp, NACT, D_MODEL, D_MODEL, 1.0f, nullptr);
  // 5) scores = (Q @ K^T) / sqrt(D) * conf  -> written straight into attn region
  const float scale = 1.0f / sqrtf((float)D_MODEL);
  gemm_nt_wmma_kernel<true, false, true>
      <<<dim3(NACT / 64, B_ROWS / 64), blk, 0, stream>>>(
      Qp, Kp, out_attn, nullptr, B_ROWS, NACT, D_MODEL, scale, conf_a);
  // 6) top-k mask + softmax in place; basin ids
  topk_softmax_kernel<<<B_ROWS, 256, NACT * sizeof(float), stream>>>(
      out_attn, active_idx, topk_ptr, out_basin, basin_i, NACT);
  // 7) ctx = attn @ V  (sparse, deterministic), bf16 out
  ctx_sparse_kernel<<<B_ROWS, 256, NACT * sizeof(float), stream>>>(
      out_attn, Vp, ctx_b, NACT, D_MODEL);
  // 8) output = ctx @ Wo^T
  gemm_nt_wmma_kernel<true, false, false>
      <<<dim3(D_MODEL / 64, B_ROWS / 64), blk, 0, stream>>>(
      ctx_b, Wo_b, out_y, nullptr, B_ROWS, D_MODEL, D_MODEL, 1.0f, nullptr);
  // 9) new_last_activated = last_activated; [basin]=NOW
  copy_f32_kernel<<<(NSLOT + 255) / 256, 256, 0, stream>>>(last_act, out_last, NSLOT);
  scatter_now_kernel<<<(B_ROWS + 255) / 256, 256, 0, stream>>>(basin_i, out_last, B_ROWS);
}
